// OneStepDecoder_82566451299214
// MI455X (gfx1250) — matrix operand order repaired and checked by reading, not compiled, not run on hardware
//
#include <hip/hip_runtime.h>
#include <hip/hip_bf16.h>
#include <stdint.h>

// ---------------- CDNA5 WMMA types ----------------
typedef __attribute__((ext_vector_type(16))) __bf16       v16bf;
typedef __attribute__((ext_vector_type(8)))  float        v8f;
typedef __attribute__((ext_vector_type(4)))  unsigned int v4u;
typedef __attribute__((ext_vector_type(4)))  float        v4f;

union FragBF16 { v4u q[2]; unsigned int u[8]; v16bf v; };

// ---------------- problem dims ----------------
#define BATCH   512
#define TENC    196
#define ENCD    1024
#define HDIM    256
#define ADIM    64
#define EMBD    300
#define NVOCAB  32000
#define GRUIN   1324
#define GRUINP  1344     // 1324 padded to multiple of 32
#define H3      768
#define ROWSTR  1028     // enc LDS tile row stride (floats): 4-bank row shift, 16B aligned
#define NTILES  13       // ceil(196/16)

// ---- CDNA5 async global->LDS path (probe-guarded) ----
#if defined(__has_builtin)
#  if __has_builtin(__builtin_amdgcn_global_load_async_to_lds_b128) && \
      __has_builtin(__builtin_amdgcn_s_wait_asynccnt)
#    define USE_ASYNC_LDS 1
#  endif
#endif
#ifndef USE_ASYNC_LDS
#  define USE_ASYNC_LDS 0
#endif

#if USE_ASYNC_LDS
// builtin expects pointers to a 16-byte int vector (per hipcc diagnostic):
//   param 1: global (__device__) address space, param 2: LDS (__shared__)
typedef int v4i_t __attribute__((vector_size(16)));
typedef v4i_t __attribute__((address_space(1)))* g_v4i_p;
typedef v4i_t __attribute__((address_space(3)))* l_v4i_p;
#  define WAIT_ASYNC(n) __builtin_amdgcn_s_wait_asynccnt(n)
#else
#  define WAIT_ASYNC(n) ((void)0)
#endif

__device__ __forceinline__ unsigned short f2bf(float f) {
    union { float f; unsigned int u; } c; c.f = f;
    unsigned int u = c.u;
    unsigned int r = u + 0x7FFFu + ((u >> 16) & 1u);   // round-to-nearest-even
    return (unsigned short)(r >> 16);
}
__device__ __forceinline__ unsigned int packbf(float a, float b) {
    return (unsigned int)f2bf(a) | ((unsigned int)f2bf(b) << 16);
}

// ---------------------------------------------------------------------------
// Transpose + convert: src f32 [K][N] -> dst bf16 [N][Kpad], zero K-padding.
// ---------------------------------------------------------------------------
__global__ void __launch_bounds__(256)
tconv_kernel(const float* __restrict__ src, unsigned short* __restrict__ dst,
             int K, int N, int Kpad) {
    __shared__ float tile[32][33];
    int k0 = blockIdx.x * 32, n0 = blockIdx.y * 32;
    int tx = threadIdx.x & 31;
    int ty = threadIdx.x >> 5;
#pragma unroll
    for (int r = 0; r < 4; ++r) {
        int k = k0 + ty * 4 + r, n = n0 + tx;
        float v = 0.f;
        if (k < K && n < N) v = src[(size_t)k * N + n];
        tile[ty * 4 + r][tx] = v;
    }
    __syncthreads();
#pragma unroll
    for (int r = 0; r < 4; ++r) {
        int n = n0 + ty * 4 + r, k = k0 + tx;
        if (n < N && k < Kpad) dst[(size_t)n * Kpad + k] = f2bf(tile[tx][ty * 4 + r]);
    }
}

__global__ void __launch_bounds__(256)
dconv_kernel(const float* __restrict__ src, unsigned short* __restrict__ dst, int n) {
    int i = blockIdx.x * 256 + threadIdx.x;
    if (i < n) dst[i] = f2bf(src[i]);
}

// ---------------------------------------------------------------------------
// Generic wave-tile bf16 WMMA GEMM:
//   C[M][N] f32 = A[M][Kpad] bf16 * BT[N][Kpad] bf16 + bias[N]
// One wave per 16x64 tile, 4 waves / block.
// ---------------------------------------------------------------------------
__global__ void __launch_bounds__(128)
gemm_bf16_kernel(const unsigned short* __restrict__ A,
                 const unsigned short* __restrict__ BT,
                 const float* __restrict__ bias,
                 float* __restrict__ C,
                 int M, int N, int Kpad) {
    int lane = threadIdx.x & 31;
    int wid  = threadIdx.x >> 5;
    int tiles_n = N >> 6;
    int id = blockIdx.x * 4 + wid;
    int mt = id / tiles_n;
    int nt = id % tiles_n;
    if (mt * 16 >= M) return;

    int half = lane >> 4;
    int l15  = lane & 15;

    const unsigned short* arow = A + (size_t)(mt * 16 + l15) * Kpad;
    v8f acc[4] = {};

    for (int kk = 0; kk < Kpad; kk += 32) {
        FragBF16 af;
        af.q[0] = *(const v4u*)(arow + kk + half * 8);
        af.q[1] = *(const v4u*)(arow + kk + 16 + half * 8);
#pragma unroll
        for (int s = 0; s < 4; ++s) {
            int col = nt * 64 + s * 16 + l15;
            const unsigned short* bcol = BT + (size_t)col * Kpad + kk + half * 8;
            FragBF16 bf_;
            bf_.q[0] = *(const v4u*)(bcol);
            bf_.q[1] = *(const v4u*)(bcol + 16);
            acc[s] = __builtin_amdgcn_wmma_f32_16x16x32_bf16(
                false, af.v, false, bf_.v, (short)0, acc[s], false, false);
        }
    }

    int rowbase = mt * 16 + half * 8;
#pragma unroll
    for (int s = 0; s < 4; ++s) {
        int col = nt * 64 + s * 16 + l15;
        float bb = bias[col];
#pragma unroll
        for (int j = 0; j < 8; ++j) {
            C[(size_t)(rowbase + j) * N + col] = acc[s][j] + bb;
        }
    }
}

// ---------------------------------------------------------------------------
// Tile stage helper: 16x1024 f32 rows -> LDS buffer (zero-fill invalid rows).
// Async path: GLOBAL_LOAD_ASYNC_TO_LDS_B128 (ASYNCcnt); else direct copy.
// 16 issues per thread (=> 16 per wave per tile; <=32 outstanding < 63 max).
// ---------------------------------------------------------------------------
__device__ __forceinline__ void issue_tile(const float* __restrict__ encb, int t0,
                                           int nvalid, float* buf, int tid) {
    for (int i = tid; i < 16 * 256; i += 256) {
        int tr = i >> 8;
        int dc = (i & 255) * 4;
        float* lp = buf + tr * ROWSTR + dc;
        if (tr < nvalid) {
            const float* gp = encb + (size_t)(t0 + tr) * ENCD + dc;
#if USE_ASYNC_LDS
            __builtin_amdgcn_global_load_async_to_lds_b128(
                (g_v4i_p)(float*)gp, (l_v4i_p)lp, 0, 0);
#else
            *(v4f*)lp = *(const v4f*)gp;
#endif
        } else {
            v4f z = {};
            *(v4f*)lp = z;
        }
    }
}

// ---------------------------------------------------------------------------
// Fused attention: score (WMMA) + online softmax + context, single pass over enc
// with double-buffered (async) LDS staging.
// One workgroup per batch row b; 256 threads = 8 waves.
// Wave w: N-subtile (w&3), K-half (w>>2) of the 16x64 score tile, K=1024.
// ---------------------------------------------------------------------------
__global__ void __launch_bounds__(256)
attn_fused_kernel(const float* __restrict__ enc,          // [B][T][1024]
                  const unsigned short* __restrict__ W2t, // [64][1024] bf16
                  const float* __restrict__ hproj,        // [B][64] (incl. b1)
                  const float* __restrict__ b2,           // [64]
                  const float* __restrict__ Wv,           // [64]
                  const float* __restrict__ bv,           // [1]
                  float* __restrict__ att,                // [B][T]
                  float* __restrict__ ctx)                // [B][1024]
{
    __shared__ float enc_t[2][16 * ROWSTR];  // 2 x 65792 B, f32 tiles
    __shared__ float psc[8][16][17];         // partial 16x16 score tiles
    __shared__ float hpb[ADIM];              // hproj + b2
    __shared__ float wv[ADIM];
    __shared__ float all_sc[TENC];
    __shared__ float sc16[16];
    __shared__ float ev[16];
    __shared__ float mrun_s, srun_s, alpha_s;

    int b    = blockIdx.x;
    int tid  = threadIdx.x;
    int lane = tid & 31, w = tid >> 5;
    int nsub = w & 3, khalf = w >> 2;
    int half = lane >> 4, l15 = lane & 15;

    if (tid < ADIM) { hpb[tid] = hproj[b * ADIM + tid] + b2[tid]; wv[tid] = Wv[tid]; }
    if (tid == 0)   { mrun_s = -3.4e38f; srun_s = 0.f; }

    float c0 = 0.f, c1 = 0.f, c2 = 0.f, c3 = 0.f;
    const float* encb = enc + (size_t)b * TENC * ENCD;
    float bvv = bv[0];

    // prologue: issue tile 0 into buffer 0
    issue_tile(encb, 0, 16, enc_t[0], tid);

    for (int ti = 0; ti < NTILES; ++ti) {
        int t0 = ti * 16;
        int nvalid = TENC - t0; if (nvalid > 16) nvalid = 16;
        float* buf = enc_t[ti & 1];

        if (ti + 1 < NTILES) {
            __syncthreads();               // prior reads of other buffer finished
            int t0n = t0 + 16;
            int nvn = TENC - t0n; if (nvn > 16) nvn = 16;
            issue_tile(encb, t0n, nvn, enc_t[(ti + 1) & 1], tid);
            WAIT_ASYNC(16);                // in-order: first 16 (tile ti) landed
        } else {
            WAIT_ASYNC(0);
        }
        __syncthreads();                   // tile ti visible block-wide

        // ---- per-wave WMMA partial scores: 16 rows x 16 cols over K=512 ----
        v8f acc = {};
        const float* arow = buf + l15 * ROWSTR + khalf * 512;
        const unsigned short* bbase =
            W2t + (size_t)(nsub * 16 + l15) * ENCD + khalf * 512;
        for (int kk = 0; kk < 512; kk += 32) {
            FragBF16 af;
            int ka = kk + half * 8;
#pragma unroll
            for (int i = 0; i < 4; ++i) {
                af.u[i]     = packbf(arow[ka + 2 * i],      arow[ka + 2 * i + 1]);
                af.u[4 + i] = packbf(arow[ka + 16 + 2 * i], arow[ka + 16 + 2 * i + 1]);
            }
            FragBF16 bf_;
            bf_.q[0] = *(const v4u*)(bbase + kk + half * 8);
            bf_.q[1] = *(const v4u*)(bbase + kk + 16 + half * 8);
            acc = __builtin_amdgcn_wmma_f32_16x16x32_bf16(
                false, af.v, false, bf_.v, (short)0, acc, false, false);
        }
#pragma unroll
        for (int j = 0; j < 8; ++j)
            psc[w][half * 8 + j][l15] = acc[j];
        __syncthreads();

        // ---- combine K-halves + tanh epilogue: 16 threads, one row each ----
        if (tid < 16) {
            float ssum = 0.f;
            for (int n = 0; n < ADIM; ++n) {
                int ns = n >> 4, nl = n & 15;
                float v = psc[ns][tid][nl] + psc[4 + ns][tid][nl];
                ssum += wv[n] * tanhf(v + hpb[n]);
            }
            float scv = ssum + bvv;
            sc16[tid] = scv;
            if (tid < nvalid) all_sc[t0 + tid] = scv;
        }
        __syncthreads();

        // ---- online softmax update (thread 0) ----
        if (tid == 0) {
            float m = mrun_s, tm = m;
            for (int t = 0; t < nvalid; ++t) tm = fmaxf(tm, sc16[t]);
            float al = __expf(m - tm);
            float s = srun_s * al;
            for (int t = 0; t < 16; ++t) {
                float e = (t < nvalid) ? __expf(sc16[t] - tm) : 0.f;
                ev[t] = e; s += e;
            }
            mrun_s = tm; srun_s = s; alpha_s = al;
        }
        __syncthreads();

        // ---- rescale + accumulate context (4 d-columns per thread) ----
        float al = alpha_s;
        c0 *= al; c1 *= al; c2 *= al; c3 *= al;
        for (int t = 0; t < 16; ++t) {
            float e = ev[t];
            const float* er = buf + t * ROWSTR;
            c0 += e * er[tid];
            c1 += e * er[tid + 256];
            c2 += e * er[tid + 512];
            c3 += e * er[tid + 768];
        }
    }
    __syncthreads();

    float inv = 1.f / srun_s;
    float m   = mrun_s;
    float* cb = ctx + (size_t)b * ENCD;
    cb[tid]       = c0 * inv;
    cb[tid + 256] = c1 * inv;
    cb[tid + 512] = c2 * inv;
    cb[tid + 768] = c3 * inv;
    for (int t = tid; t < TENC; t += 256)
        att[b * TENC + t] = __expf(all_sc[t] - m) * inv;
}

// ---------------------------------------------------------------------------
// x_bf16[b][0..1343] = bf16( concat(context[b], emb_table[idx[b]], zeros[20]) )
// ---------------------------------------------------------------------------
__global__ void __launch_bounds__(256)
buildx_kernel(const float* __restrict__ ctx, const float* __restrict__ emb,
              const int* __restrict__ idx, unsigned short* __restrict__ xbf) {
    int b = blockIdx.x;
    int row = idx[b];
    for (int j = threadIdx.x; j < GRUINP; j += 256) {
        float v;
        if (j < ENCD)             v = ctx[b * ENCD + j];
        else if (j < ENCD + EMBD) v = emb[(size_t)row * EMBD + (j - ENCD)];
        else                      v = 0.f;
        xbf[(size_t)b * GRUINP + j] = f2bf(v);
    }
}

// ---------------------------------------------------------------------------
// GRU gate fusion (reset_after=True). grid = B, block = 256 (= H).
// ---------------------------------------------------------------------------
__global__ void __launch_bounds__(256)
gru_gate_kernel(const float* __restrict__ mx, const float* __restrict__ mi,
                const float* __restrict__ h_old,
                float* __restrict__ h_new, unsigned short* __restrict__ h_new_bf) {
    int b = blockIdx.x, n = threadIdx.x;
    float xz = mx[b * H3 + n],            iz = mi[b * H3 + n];
    float xr = mx[b * H3 + HDIM + n],     ir = mi[b * H3 + HDIM + n];
    float xh = mx[b * H3 + 2 * HDIM + n], ih = mi[b * H3 + 2 * HDIM + n];
    float z = 1.f / (1.f + __expf(-(xz + iz)));
    float r = 1.f / (1.f + __expf(-(xr + ir)));
    float hh = tanhf(xh + r * ih);
    float h  = z * h_old[b * HDIM + n] + (1.f - z) * hh;
    h_new[b * HDIM + n]    = h;
    h_new_bf[b * HDIM + n] = f2bf(h);
}

// ---------------------------------------------------------------------------
extern "C" void kernel_launch(void* const* d_in, const int* in_sizes, int n_in,
                              void* d_out, int out_size, void* d_ws, size_t ws_size,
                              hipStream_t stream) {
    const int*   d_idx = (const int*)  d_in[0];
    const float* d_enc = (const float*)d_in[1];
    const float* d_h   = (const float*)d_in[2];
    const float* d_emb = (const float*)d_in[3];
    const float* d_W1  = (const float*)d_in[4];
    const float* d_b1  = (const float*)d_in[5];
    const float* d_W2  = (const float*)d_in[6];
    const float* d_b2  = (const float*)d_in[7];
    const float* d_Wv  = (const float*)d_in[8];
    const float* d_bv  = (const float*)d_in[9];
    const float* d_gk  = (const float*)d_in[10];
    const float* d_grk = (const float*)d_in[11];
    const float* d_gib = (const float*)d_in[12];
    const float* d_grb = (const float*)d_in[13];
    const float* d_Wd  = (const float*)d_in[14];
    const float* d_bd  = (const float*)d_in[15];

    // outputs (concatenated flat, f32): logits, h_new, att_weights, context
    float* out_logits = (float*)d_out;
    float* out_hnew   = out_logits + (size_t)BATCH * NVOCAB;
    float* out_att    = out_hnew   + (size_t)BATCH * HDIM;
    float* out_ctx    = out_att    + (size_t)BATCH * TENC;

    // workspace layout (byte offsets, 256-aligned)
    uint8_t* w = (uint8_t*)d_ws;
    float*          hproj  = (float*)         (w + 0);         // 512*64*4
    unsigned short* W1t    = (unsigned short*)(w + 532480);    // 64*256*2
    unsigned short* W2t    = (unsigned short*)(w + 565248);    // 64*1024*2
    unsigned short* gruKt  = (unsigned short*)(w + 696320);    // 768*1344*2
    unsigned short* recKt  = (unsigned short*)(w + 2760704);   // 768*256*2
    unsigned short* WdT    = (unsigned short*)(w + 3153920);   // 32000*256*2
    unsigned short* h_bf   = (unsigned short*)(w + 19537920);  // 512*256*2
    unsigned short* x_bf   = (unsigned short*)(w + 19800064);  // 512*1344*2
    float*          mx     = (float*)         (w + 21176320);  // 512*768*4
    float*          mi     = (float*)         (w + 22749184);  // 512*768*4
    unsigned short* hnewbf = (unsigned short*)(w + 24322048);  // 512*256*2

    // --- weight conversion (f32 -> bf16, N-major, K padded) ---
    tconv_kernel<<<dim3(8, 2),    256, 0, stream>>>(d_W1,  W1t,   HDIM,  ADIM,   HDIM);
    tconv_kernel<<<dim3(32, 2),   256, 0, stream>>>(d_W2,  W2t,   ENCD,  ADIM,   ENCD);
    tconv_kernel<<<dim3(42, 24),  256, 0, stream>>>(d_gk,  gruKt, GRUIN, H3,     GRUINP);
    tconv_kernel<<<dim3(8, 24),   256, 0, stream>>>(d_grk, recKt, HDIM,  H3,     HDIM);
    tconv_kernel<<<dim3(8, 1000), 256, 0, stream>>>(d_Wd,  WdT,   HDIM,  NVOCAB, HDIM);
    dconv_kernel<<<512, 256, 0, stream>>>(d_h, h_bf, BATCH * HDIM);

    // --- h_proj = state_h @ W1 + b1 : M=512 N=64 K=256 ---
    gemm_bf16_kernel<<<8, 128, 0, stream>>>(h_bf, W1t, d_b1, hproj, BATCH, ADIM, HDIM);

    // --- fused attention: score(WMMA) + online softmax + context, 1 pass over enc ---
    attn_fused_kernel<<<BATCH, 256, 0, stream>>>(d_enc, W2t, hproj, d_b2, d_Wv, d_bv,
                                                 out_att, out_ctx);

    // --- GRU input x = [context, embedding], bf16, padded ---
    buildx_kernel<<<BATCH, 256, 0, stream>>>(out_ctx, d_emb, d_idx, x_bf);

    // --- mx = x @ gru_kernel + bias : M=512 N=768 K=1344 ---
    gemm_bf16_kernel<<<96, 128, 0, stream>>>(x_bf, gruKt, d_gib, mx, BATCH, H3, GRUINP);
    // --- mi = h @ gru_rec_kernel + bias : M=512 N=768 K=256 ---
    gemm_bf16_kernel<<<96, 128, 0, stream>>>(h_bf, recKt, d_grb, mi, BATCH, H3, HDIM);

    // --- GRU gates -> h_new (f32 out + bf16 for logits GEMM) ---
    gru_gate_kernel<<<BATCH, 256, 0, stream>>>(mx, mi, d_h, out_hnew, hnewbf);

    // --- logits = h_new @ Wd + bd : M=512 N=32000 K=256 ---
    gemm_bf16_kernel<<<4000, 128, 0, stream>>>(hnewbf, WdT, d_bd, out_logits,
                                               BATCH, NVOCAB, HDIM);
}
